// EnhancedHungarianMatcher_86346022518894
// MI455X (gfx1250) — compile-verified, matrix operand order checked by reading, not run on hardware
//
#include <hip/hip_runtime.h>
#include <hip/hip_bf16.h>

typedef __attribute__((ext_vector_type(16))) _Float16 v16h;
typedef __attribute__((ext_vector_type(8)))  float    v8f;
typedef __attribute__((ext_vector_type(4)))  unsigned int u32x4;
typedef __attribute__((ext_vector_type(8)))  int      i32x8;
typedef __attribute__((ext_vector_type(4)))  int      i32x4;

#define Bn 8
#define Qn 200
#define Cn 20
#define Pn 30000
#define QT 13              // ceil(200/16)
#define CHUNK 3840         // cols per workgroup
#define NCH 8              // ceil(30000/3840)
#define HBINS 2048
#define TCOLS 256          // cols per TDM tile
#define NTILES (CHUNK / TCOLS)   // 15
#define TROWSTR 260        // LDS row stride in floats (256 + 4-dword TDM pad)
#define TILEB (16 * TROWSTR * 4) // 16640 bytes per tile buffer

// dynamic LDS layout (bytes)
#define OFF_PRED 0                        // 2 buffers
#define OFF_GT   (2 * TILEB)              // 2 buffers
#define OFF_HC   (4 * TILEB)              // HBINS u32
#define OFF_HP   (OFF_HC + HBINS * 4)     // HBINS f32
#define SMEM_BYTES (OFF_HP + HBINS * 4)   // 82944

// workspace word (f32/i32) layout
#define W_SP   0
#define W_SPQ  1600
#define W_SM   3200
#define W_SMT  4800
#define W_ST   6400
#define W_BCE  8000
#define W_POS  8008
#define W_CMIN 8024
#define W_CMAX 8032
#define W_SCAL 8040
#define W_HIST 8192
#define W_TOT  (W_HIST + 2 * Bn * HBINS)  // 40960 words

__global__ void __launch_bounds__(256) hm_init(float* ws) {
  int i = blockIdx.x * 256 + threadIdx.x;
  if (i < W_TOT) {
    int v = 0;
    if (i >= W_CMIN && i < W_CMIN + Bn) v = 0x7fffffff;
    else if (i >= W_CMAX && i < W_CMAX + Bn) v = -1;
    ((int*)ws)[i] = v;
  }
}

// Issue one TDM 2D-tile load: 16 rows x TCOLS elems of 4-byte data -> LDS,
// row-padded by 4 DWORDs per 256 DWORDs (pad_interval=7, pad_amount code 3).
// OOB global reads (x >= td0 or y >= td1) zero-fill LDS per ISA 8.2.
__device__ __forceinline__ void tdm_issue(const void* gbase, long elem_off,
                                          int td0, int td1, unsigned lds_off) {
  const unsigned long long ga =
      (unsigned long long)gbase + (unsigned long long)elem_off * 4ull;
  u32x4 g0;
  g0[0] = 1u;                                       // count=1, user mode
  g0[1] = lds_off;                                  // lds_addr (bytes)
  g0[2] = (unsigned)ga;                             // global_addr[31:0]
  g0[3] = (unsigned)((ga >> 32) & 0x1FFFFFFull) | (2u << 30); // addr[56:32]|type=2
  i32x8 g1;
  g1[0] = (int)((2u << 16) | (1u << 20) | (7u << 22) | (3u << 25));
  //            data_size=4B  pad_en      interval=256dw  amount=4dw
  g1[1] = (int)(((unsigned)td0 & 0xFFFFu) << 16);   // abar=0 | tensor_dim0 lo
  g1[2] = (int)(((unsigned)td0 >> 16) | (((unsigned)td1 & 0xFFFFu) << 16));
  g1[3] = (int)(((unsigned)td1 >> 16) | ((unsigned)TCOLS << 16)); // tile_dim0
  g1[4] = 16;                                       // tile_dim1=16, tile_dim2=0
  g1[5] = Pn;                                       // tensor_dim0_stride lo
  g1[6] = 0;                                        // stride0 hi | stride1 lo
  g1[7] = 0;
  const i32x4 z4 = {};
  const i32x8 z8 = {};
  __builtin_amdgcn_tensor_load_to_lds(g0, g1, z4, z4, z8, 0);
}

__global__ void __launch_bounds__(256) hm_main(const float* __restrict__ pm,
                                               const int* __restrict__ gm,
                                               float* __restrict__ ws) {
  extern __shared__ char smem[];
  const int chunk = blockIdx.x, qt = blockIdx.y, b = blockIdx.z;
  const int tid  = threadIdx.x;
  const int lane = tid & 31;
  const int wave = tid >> 5;
  const int row  = lane & 15;            // A-fragment: lane L supplies row (L&15)
  const int hlf  = lane >> 4;
  const int q    = qt * 16 + row;
  const bool qv  = (q < Qn);

  unsigned* hc = (unsigned*)(smem + OFF_HC);
  float*    hp = (float*)(smem + OFF_HP);
  for (int i = tid; i < HBINS; i += 256) { hc[i] = 0u; hp[i] = 0.f; }

  const unsigned lds_base = (unsigned)(unsigned long long)(void*)smem;
  const long tile_row0 = (long)(b * Qn + qt * 16) * Pn;
  const int td1 = Qn - qt * 16;          // rows available; <16 only for qt=12

  // prologue: issue tile 0 (pred + gt)
  if (wave == 0) {
    const int c0 = chunk * CHUNK;
    tdm_issue(pm, tile_row0 + c0, Pn - c0, td1, lds_base + OFF_PRED);
    tdm_issue(gm, tile_row0 + c0, Pn - c0, td1, lds_base + OFF_GT);
  }

  v16h ones;
#pragma unroll
  for (int e = 0; e < 16; ++e) ones[e] = (_Float16)1.0f;

  v8f accP = {}, accPT = {}, accM = {}, accMT = {};
  float tsum = 0.f, bces = 0.f, poss = 0.f;
  int cmn = 0x7fffffff, cmx = -1;

  for (int i = 0; i < NTILES; ++i) {
    const int buf  = i & 1;
    const int colT = chunk * CHUNK + i * TCOLS;         // tile base column
    const int colN = colT + TCOLS;                      // next tile base
    if (wave == 0) {
      if (i + 1 < NTILES && colN < Pn) {
        tdm_issue(pm, tile_row0 + colN, Pn - colN, td1,
                  lds_base + OFF_PRED + ((i + 1) & 1) * TILEB);
        tdm_issue(gm, tile_row0 + colN, Pn - colN, td1,
                  lds_base + OFF_GT + ((i + 1) & 1) * TILEB);
        __builtin_amdgcn_s_wait_tensorcnt((short)2);    // tile i complete
      } else {
        __builtin_amdgcn_s_wait_tensorcnt((short)0);
      }
    }
    __syncthreads();                                    // tile i visible to all

    if (colT < Pn) {
      const int csub = wave * 32 + hlf * 16;            // this lane's 16 cols
      const char* lp = smem + OFF_PRED + buf * TILEB + (row * TROWSTR + csub) * 4;
      const char* lg = smem + OFF_GT   + buf * TILEB + (row * TROWSTR + csub) * 4;
      float xs[16], tv[16];
#pragma unroll
      for (int k = 0; k < 4; ++k) {
        const float4 fx = ((const float4*)lp)[k];
        const int4   gi = ((const int4*)lg)[k];
        xs[4 * k + 0] = fx.x; xs[4 * k + 1] = fx.y;
        xs[4 * k + 2] = fx.z; xs[4 * k + 3] = fx.w;
        tv[4 * k + 0] = (float)gi.x; tv[4 * k + 1] = (float)gi.y;
        tv[4 * k + 2] = (float)gi.z; tv[4 * k + 3] = (float)gi.w;
      }

      v16h fP, fPT, fM, fMT;
#pragma unroll
      for (int e = 0; e < 16; ++e) {
        const int p = colT + csub + e;
        const bool v = qv && (p < Pn);
        const float x = xs[e], t = tv[e];
        const float u  = __expf(-x);
        const float sg = 1.f / (1.f + u);               // sigmoid(x)
        const float m2 = 1.f / (1.f + __expf(-sg));     // double sigmoid
        if (v) {
          const float l1u = log1pf(u);                  // softplus pieces
          if (t > 0.5f) {                               // ce = l1u
            const float om = 1.f - sg;
            bces += om * om * 0.25f * l1u;
            tsum += 1.f;
            poss += om;
            cmn = min(cmn, p);
            cmx = max(cmx, p);
          } else {                                      // ce = x + l1u
            bces += sg * sg * 0.75f * (x + l1u);
            int bin = (int)((8.f - x) * (HBINS / 16.f));
            bin = max(0, min(HBINS - 1, bin));
            atomicAdd(&hc[bin], 1u);
            atomicAdd(&hp[bin], sg);
          }
        }
        const float sgv = v ? sg : 0.f;
        const float m2v = v ? m2 : 0.f;
        fP[e]  = (_Float16)sgv;
        fPT[e] = (_Float16)(sgv * t);
        fM[e]  = (_Float16)m2v;
        fMT[e] = (_Float16)(m2v * t);
      }
      // B = all-ones => D[m,n] = row-sum of A row m; K-slot order irrelevant.
      accP  = __builtin_amdgcn_wmma_f32_16x16x32_f16(false, fP,  false, ones, (short)0, accP,  false, false);
      accPT = __builtin_amdgcn_wmma_f32_16x16x32_f16(false, fPT, false, ones, (short)0, accPT, false, false);
      accM  = __builtin_amdgcn_wmma_f32_16x16x32_f16(false, fM,  false, ones, (short)0, accM,  false, false);
      accMT = __builtin_amdgcn_wmma_f32_16x16x32_f16(false, fMT, false, ones, (short)0, accMT, false, false);
    }
    __syncthreads();                     // buffer consumed before re-fill
  }

  // C/D layout: lane 0 VGPR j = row j sum, lane 16 VGPR j = row 8+j sum
  if (row == 0) {
#pragma unroll
    for (int j = 0; j < 8; ++j) {
      const int r2 = qt * 16 + j + 8 * hlf;
      if (r2 < Qn) {
        const int o = b * Qn + r2;
        atomicAdd(&ws[W_SP  + o], accP[j]);
        atomicAdd(&ws[W_SPQ + o], accPT[j]);
        atomicAdd(&ws[W_SM  + o], accM[j]);
        atomicAdd(&ws[W_SMT + o], accMT[j]);
      }
    }
  }
  if (qv) atomicAdd(&ws[W_ST + b * Qn + q], tsum);

#pragma unroll
  for (int m = 16; m > 0; m >>= 1) {
    bces += __shfl_xor(bces, m, 32);
    poss += __shfl_xor(poss, m, 32);
    cmn = min(cmn, __shfl_xor(cmn, m, 32));
    cmx = max(cmx, __shfl_xor(cmx, m, 32));
  }
  if (lane == 0) {
    atomicAdd(&ws[W_BCE + b], bces);
    atomicAdd(&ws[W_POS + b], poss);
    atomicMin((int*)ws + W_CMIN + b, cmn);
    atomicMax((int*)ws + W_CMAX + b, cmx);
  }
  __syncthreads();
  unsigned* Hc = (unsigned*)ws + W_HIST + b * HBINS;
  float*    Hp = ws + W_HIST + Bn * HBINS + b * HBINS;
  for (int i = tid; i < HBINS; i += 256) {
    if (hc[i]) {
      atomicAdd(&Hc[i], hc[i]);
      atomicAdd(&Hp[i], hp[i]);
    }
  }
}

__device__ __forceinline__ float block_sum(float v, float* red, int tid) {
  red[tid] = v; __syncthreads();
  for (int s = 128; s > 0; s >>= 1) { if (tid < s) red[tid] += red[tid + s]; __syncthreads(); }
  float r = red[0]; __syncthreads();
  return r;
}

__global__ void __launch_bounds__(256) hm_finalize(const float* __restrict__ cw,
                                                   float* __restrict__ ws) {
  const int b = blockIdx.x, tid = threadIdx.x;
  __shared__ float Rr[HBINS + 1];
  __shared__ float red[256];

  float gpart = 0.f;
  for (int q = tid; q < Qn; q += 256) gpart += ws[W_ST + b * Qn + q];
  const float g = block_sum(gpart, red, tid);          // N1 = #positives

  const unsigned* Hc = (const unsigned*)ws + W_HIST + b * HBINS;
  const float*    Hp = ws + W_HIST + Bn * HBINS + b * HBINS;
  if (tid == 0) {
    float acc = 0.f;
    for (int i = 0; i < HBINS; ++i) { Rr[i] = acc; acc += (float)Hc[i]; }
    Rr[HBINS] = acc;
  }
  __syncthreads();

  // lovasz negatives: sum_r p_(r) * g/((g+r)(g+r+1)); per-bin telescoped:
  // sum_{r=R}^{R+c-1} = c*g/((g+R)(g+R+c))  (exact, no cancellation)
  float negterm = 0.f;
  if (g > 0.f) {
    for (int i = tid; i < HBINS; i += 256) {
      const float c = (float)Hc[i];
      if (c > 0.f) {
        const float R = Rr[i];
        negterm += Hp[i] * (g / ((g + R) * (g + R + c)));
      }
    }
  }

  const int cmn = ((int*)ws)[W_CMIN + b];
  const int cmx = ((int*)ws)[W_CMAX + b];
  const bool hasany = (cmx >= 0);
  const float span = hasany ? (float)(cmx - cmn) : 0.f;
  const float encl = (float)(Pn - 1) * span;

  float dsum = 0.f, gsum = 0.f;
  for (int q = tid; q < Qn; q += 256) {
    const int o = b * Qn + q;
    const float sp  = ws[W_SP + o],  spq = ws[W_SPQ + o];
    const float sm  = ws[W_SM + o],  smt = ws[W_SMT + o];
    const float st  = ws[W_ST + o];
    dsum += 1.f - (2.f * spq + 1.f) / (sp + st + 1.f);
    const float uni = sm + st - smt;
    const float iou = smt / (uni + 1e-6f);
    if (hasany) gsum += 1.f - (iou - (encl - uni) / (encl + 1e-6f));
    else        gsum += 1.f - iou;
  }

  const float negsum = block_sum(negterm, red, tid);
  const float dtot   = block_sum(dsum, red, tid);
  const float gtot   = block_sum(gsum, red, tid);

  if (tid == 0) {
    const float Ntot = (float)Qn * (float)Pn;
    const float N0   = Ntot - g;
    const float lov  = (ws[W_POS + b] + N0) / Ntot + negsum;
    const float bce  = ws[W_BCE + b] / (Ntot * (float)Pn);
    const float dice = dtot / (float)Qn;
    const float giou = gtot / (float)Qn;
    ws[W_SCAL + b] = cw[1] * bce + cw[2] * dice + cw[3] * giou + cw[4] * lov;
  }
}

__global__ void __launch_bounds__(256) hm_emit(const float* __restrict__ pl,
                                               const int* __restrict__ gl,
                                               const float* __restrict__ cw,
                                               const float* __restrict__ ws,
                                               float* __restrict__ out) {
  const int bq = blockIdx.x;          // b*Qn + q
  const int b = bq / Qn;
  const int tid = threadIdx.x;
  __shared__ float prob[Cn];
  if (tid == 0) {
    float l[Cn], mx = -1e30f;
    for (int c = 0; c < Cn; ++c) { l[c] = pl[(long)bq * Cn + c]; mx = fmaxf(mx, l[c]); }
    float den = 0.f;
    for (int c = 0; c < Cn; ++c) { l[c] = __expf(l[c] - mx); den += l[c]; }
    for (int c = 0; c < Cn; ++c) prob[c] = l[c] / den;
  }
  __syncthreads();
  const float w0 = cw[0];
  const float sc = ws[W_SCAL + b];
  for (int tt = tid; tt < Qn; tt += 256) {
    const int cls = gl[b * Qn + tt];
    out[(long)bq * Qn + tt] = -w0 * prob[cls] + sc;
  }
}

extern "C" void kernel_launch(void* const* d_in, const int* in_sizes, int n_in,
                              void* d_out, int out_size, void* d_ws, size_t ws_size,
                              hipStream_t stream) {
  const float* pred_labels = (const float*)d_in[0];   // [B,Q,C]
  const float* pred_masks  = (const float*)d_in[1];   // [B,Q,P]
  const float* cost_weight = (const float*)d_in[2];   // [5]
  const int*   gt_labels   = (const int*)d_in[3];     // [B,Q]
  const int*   gt_masks    = (const int*)d_in[4];     // [B,Q,P]
  float* ws  = (float*)d_ws;
  float* out = (float*)d_out;
  if (ws_size < (size_t)W_TOT * sizeof(float)) return;

  hm_init<<<(W_TOT + 255) / 256, 256, 0, stream>>>(ws);
  dim3 grid(NCH, QT, Bn);
  hm_main<<<grid, 256, SMEM_BYTES, stream>>>(pred_masks, gt_masks, ws);
  hm_finalize<<<Bn, 256, 0, stream>>>(cost_weight, ws);
  hm_emit<<<Bn * Qn, 256, 0, stream>>>(pred_labels, gt_labels, cost_weight, ws, out);
}
